// Attention_36962488549574
// MI455X (gfx1250) — compile-verified, matrix operand order checked by reading, not run on hardware
//
#include <hip/hip_runtime.h>

// ---------------------------------------------------------------------------
// Problem constants: B=2, N=2048, DIM=4096, N_HEAD=32, N_KV_HEAD=8, HEAD_DIM=128
// ---------------------------------------------------------------------------
#define SEQ      2048
#define BATCH    2
#define DIM      4096
#define NH       32
#define NKVH     8
#define HD       128
#define MROWS    (BATCH * SEQ)      // 4096 flattened token rows
#define KVDIM    (NKVH * HD)        // 1024

typedef _Float16 f16;
typedef __attribute__((ext_vector_type(16))) _Float16 v16h;
typedef __attribute__((ext_vector_type(8)))  _Float16 v8h;
typedef __attribute__((ext_vector_type(8)))  float    v8f;
typedef __attribute__((ext_vector_type(4)))  int      v4i;

// D = A(16x32) * B(32x16) + C, f16 inputs, f32 accumulate.
__device__ inline v8f wmma_f16(v16h a, v16h b, v8f c) {
    return __builtin_amdgcn_wmma_f32_16x16x32_f16(
        /*neg_a=*/false, a, /*neg_b=*/false, b,
        /*c_mod=*/(short)0, c, /*reuse_a=*/false, /*reuse_b=*/false);
}

// 16-bit operand fragment: lane holds K {koff..koff+7, koff+16..koff+23},
// koff = (lane>=16 ? 8 : 0). Caller passes pointers already offset by koff.
__device__ inline v16h make_frag(const f16* p0, const f16* p1) {
    v8h lo = *(const v8h*)p0;
    v8h hi = *(const v8h*)p1;
    v16h r;
#pragma unroll
    for (int i = 0; i < 8; ++i) { r[i] = lo[i]; r[i + 8] = hi[i]; }
    return r;
}

// ---------------------------------------------------------------------------
// CDNA5 async global -> LDS copy (16 bytes), tracked by ASYNCcnt.
// ---------------------------------------------------------------------------
typedef __attribute__((address_space(1))) v4i* global_v4i_p;
typedef __attribute__((address_space(3))) v4i* shared_v4i_p;

__device__ inline void async_copy_b128(const f16* g, f16* l) {
#if __has_builtin(__builtin_amdgcn_global_load_async_to_lds_b128)
    __builtin_amdgcn_global_load_async_to_lds_b128(
        (global_v4i_p)g, (shared_v4i_p)l, 0, 0);
#else
    unsigned loff = (unsigned)(unsigned long long)
        (__attribute__((address_space(3))) void*)l;
    asm volatile("global_load_async_to_lds_b128 %0, %1, off"
                 :: "v"(loff), "v"(g) : "memory");
#endif
}

// Wait until at most N async ops remain (async loads complete in order).
template <int N>
__device__ inline void wait_async() {
#if __has_builtin(__builtin_amdgcn_s_wait_asynccnt)
    __builtin_amdgcn_s_wait_asynccnt(N);
#else
    asm volatile("s_wait_asynccnt %0" :: "i"(N) : "memory");
#endif
}

__device__ inline void wait_ds0() {
    asm volatile("s_wait_dscnt 0x0" ::: "memory");
}

// ---------------------------------------------------------------------------
// f32 -> f16 elementwise conversion
// ---------------------------------------------------------------------------
__global__ void f32_to_f16_kernel(const float* __restrict__ in,
                                  f16* __restrict__ out, int n) {
    int i = blockIdx.x * blockDim.x + threadIdx.x;
    if (i < n) out[i] = (f16)in[i];
}

// ---------------------------------------------------------------------------
// V transpose: in = f32 (B, N, NKVH, HD); out = f16 (B, NKVH, HD, N).
// Keys become the contiguous axis so flash-attention B-fragments for P@V
// are plain 16-byte loads instead of 16-way scalar gathers.
// ---------------------------------------------------------------------------
__global__ void transpose_v_kernel(const float* __restrict__ in,
                                   f16* __restrict__ out, int total) {
    int i = blockIdx.x * blockDim.x + threadIdx.x;
    if (i >= total) return;
    const int n   = i & (SEQ - 1);
    int t         = i >> 11;            // SEQ = 2048
    const int d   = t & (HD - 1);
    t >>= 7;                            // HD = 128
    const int kvh = t & (NKVH - 1);
    const int b   = t >> 3;             // NKVH = 8
    out[i] = (f16)in[((size_t)(b * SEQ + n) * NKVH + kvh) * HD + d];
}

// ---------------------------------------------------------------------------
// C(MxN) = A(MxK) @ Bw(NxK)^T, f16 operands row-major.
// Block: 256 threads (8 waves), block tile 128x64, K-step 32.
// Double-buffered async staging of A(128x32) and B(64x32) tiles in LDS.
// ---------------------------------------------------------------------------
template <typename OutT>
__global__ __launch_bounds__(256)
void gemm_wmma_kernel(const f16* __restrict__ A, const f16* __restrict__ Bw,
                      OutT* __restrict__ C, int M, int N, int K) {
    __shared__ __align__(16) f16 ldsA[2][128 * 32];   // 16 KB
    __shared__ __align__(16) f16 ldsB[2][64 * 32];    //  8 KB

    const int tid  = threadIdx.x;
    const int lane = tid & 31;
    const int wave = tid >> 5;
    const int lid  = lane & 15;
    const int grp  = lane >> 4;
    const int koff = grp * 8;

    const int m_blk = blockIdx.y * 128;
    const int n0    = blockIdx.x * 64;

    auto stage = [&](int buf, int kk) {
        // A tile: 128 rows x 32 halves = 512 x 16B chunks, 2 per thread.
#pragma unroll
        for (int c = 0; c < 2; ++c) {
            const int ch  = tid + c * 256;
            const int row = ch >> 2, part = ch & 3;
            async_copy_b128(A + (size_t)(m_blk + row) * K + kk + part * 8,
                            &ldsA[buf][ch * 8]);
        }
        // B tile: 64 rows x 32 halves = 256 x 16B chunks, 1 per thread.
        {
            const int row = tid >> 2, part = tid & 3;
            async_copy_b128(Bw + (size_t)(n0 + row) * K + kk + part * 8,
                            &ldsB[buf][tid * 8]);
        }
    };

    v8f acc[4] = {};
    stage(0, 0);
    int buf = 0;
    for (int kk = 0; kk < K; kk += 32) {
        wait_async<0>();      // this wave's loads for `buf` are done
        __syncthreads();      // all waves' loads done; prev buffer fully read
        if (kk + 32 < K) stage(buf ^ 1, kk + 32);

        const f16* ar = &ldsA[buf][(wave * 16 + lid) * 32 + koff];
        v16h af = make_frag(ar, ar + 16);
#pragma unroll
        for (int t = 0; t < 4; ++t) {
            const f16* br = &ldsB[buf][(t * 16 + lid) * 32 + koff];
            acc[t] = wmma_f16(af, make_frag(br, br + 16), acc[t]);
        }
        buf ^= 1;
    }

#pragma unroll
    for (int t = 0; t < 4; ++t) {
#pragma unroll
        for (int r = 0; r < 8; ++r) {
            const int row = m_blk + wave * 16 + r + 8 * grp;
            const int col = n0 + t * 16 + lid;
            C[(size_t)row * N + col] = (OutT)acc[t][r];
        }
    }
}

// ---------------------------------------------------------------------------
// RoPE: in = f32 (B, N, nheads, 128), out = f16 same layout.
// ---------------------------------------------------------------------------
__global__ void rope_kernel(const float* __restrict__ in, f16* __restrict__ out,
                            const float* __restrict__ cosb,
                            const float* __restrict__ sinb,
                            int nheads, int total) {
    int i = blockIdx.x * blockDim.x + threadIdx.x;
    if (i >= total) return;
    const int half = i & 63;
    const int t    = i >> 6;                 // (b*SEQ + n)*nheads + h
    const int n    = (t / nheads) & (SEQ - 1);
    const size_t base = (size_t)t * HD + 2 * half;
    const float c = cosb[n * 64 + half];
    const float s = sinb[n * 64 + half];
    const float x0 = in[base], x1 = in[base + 1];
    out[base]     = (f16)(x0 * c - x1 * s);
    out[base + 1] = (f16)(x0 * s + x1 * c);
}

// ---------------------------------------------------------------------------
// Causal flash attention.
//  qh: (B, N, NH, HD) f16 roped     kh: (B, N, NKVH, HD) f16
//  vT: (B, NKVH, HD, N) f16         oh: (B, N, NH, HD) f16
// Block = 128 threads = 4 waves; each wave owns 16 query rows (64/block).
// All 4 waves share one KV head. Per k-tile (32 keys):
//   K  tile 32x128 (rows padded to 136 halves)  -> S operands, b128 LDS reads
//   V^T tile 128x32 (rows padded to 40 halves)  -> P@V operands, b128 LDS reads
// Both async-staged double-buffered; tile j+1 loads overlap tile j compute.
// Waves run a uniform trip count; tiles past a wave's causal bound are fully
// masked and numerically inert in the online softmax.
// ---------------------------------------------------------------------------
#define KVP 136
#define VTP 40
__global__ __launch_bounds__(128)
void flash_attn_kernel(const f16* __restrict__ qh, const f16* __restrict__ kh,
                       const f16* __restrict__ vT, f16* __restrict__ oh) {
    __shared__ __align__(16) f16 ldsK[2][32 * KVP];    // 17 KB
    __shared__ __align__(16) f16 ldsVT[2][HD * VTP];   // 20 KB
    __shared__ __align__(16) f16 lds_p[4][16 * 32];    //  4 KB per-wave P

    const int tid  = threadIdx.x;
    const int lane = tid & 31;
    const int wave = tid >> 5;
    const int lid  = lane & 15;
    const int grp  = lane >> 4;
    const int koff = grp * 8;

    const int bid  = blockIdx.x;
    const int qb   = bid & 31;          // 32 query blocks of 64 rows
    const int hq   = (bid >> 5) & 31;   // query head
    const int b    = bid >> 10;         // batch
    const int kvh  = hq >> 2;           // N_REP = 4
    const int m_base = qb * 64 + wave * 16;

    // Q tile (16 x 128) as 4 A-fragments.
    const f16* qbase =
        qh + (size_t)(b * SEQ + m_base + lid) * (NH * HD) + hq * HD;
    v16h aq[4];
#pragma unroll
    for (int c = 0; c < 4; ++c)
        aq[c] = make_frag(qbase + c * 32 + koff, qbase + c * 32 + koff + 16);

    v8f  o[8] = {};
    float mrow[8], lrow[8];
#pragma unroll
    for (int r = 0; r < 8; ++r) { mrow[r] = -INFINITY; lrow[r] = 0.0f; }

    f16* myp = lds_p[wave];
    const int jmax_blk = (qb * 64 + 63) >> 5;  // uniform across waves
    const float scale = 0.08838834764831845f;  // 1/sqrt(128)

    // Cooperative async stage of K + V^T tile j into buffer `buf`.
    // 512 chunks each; 128 threads -> 4+4 async ops per thread,
    // i.e. exactly 8 async instructions per wave per stage.
    const size_t vbase_g = (size_t)((b * NKVH + kvh) * HD) * SEQ;
    auto stage = [&](int buf, int j) {
        const int kbase = j * 32;
        const size_t kbase_g =
            (size_t)(b * SEQ + kbase) * (NKVH * HD) + kvh * HD;
#pragma unroll
        for (int c = 0; c < 4; ++c) {
            const int ch = tid + c * 128;
            {   // K: 32 rows x 16 chunks
                const int row = ch >> 4, part = ch & 15;
                async_copy_b128(kh + kbase_g + (size_t)row * (NKVH * HD) + part * 8,
                                &ldsK[buf][row * KVP + part * 8]);
            }
            {   // V^T: 128 rows x 4 chunks
                const int row = ch >> 2, part = ch & 3;
                async_copy_b128(vT + vbase_g + (size_t)row * SEQ + kbase + part * 8,
                                &ldsVT[buf][row * VTP + part * 8]);
            }
        }
    };

    stage(0, 0);
    int buf = 0;
    for (int j = 0; j <= jmax_blk; ++j) {
        const int kbase = j * 32;

        // Prefetch next tile, then wait for current (async completes in order:
        // cnt<=8 means tile j's 8 instructions have retired).
        if (j < jmax_blk) { stage(buf ^ 1, j + 1); wait_async<8>(); }
        else              { wait_async<0>(); }
        __syncthreads();   // all waves' stores for tile j visible

        // ---- S = Q @ K^T for 32 keys, K from LDS ----
        v8f s[2] = {};
#pragma unroll
        for (int t = 0; t < 2; ++t) {
            const f16* kb = &ldsK[buf][(t * 16 + lid) * KVP + koff];
#pragma unroll
            for (int c = 0; c < 4; ++c)
                s[t] = wmma_f16(aq[c], make_frag(kb + c * 32, kb + c * 32 + 16),
                                s[t]);
        }

        // ---- scale + causal mask (covers partially & fully masked tiles) ----
        const bool need_mask = (kbase + 31) > m_base;
#pragma unroll
        for (int t = 0; t < 2; ++t) {
#pragma unroll
            for (int r = 0; r < 8; ++r) {
                float val = s[t][r] * scale;
                if (need_mask) {
                    const int key = kbase + t * 16 + lid;
                    const int qr  = m_base + r + 8 * grp;
                    if (key > qr) val = -1.0e9f;
                }
                s[t][r] = val;
            }
        }

        // ---- online softmax (row reductions over each 16-lane group) ----
        float alpha[8];
#pragma unroll
        for (int r = 0; r < 8; ++r) {
            float lm = fmaxf(s[0][r], s[1][r]);
#pragma unroll
            for (int xm = 1; xm < 16; xm <<= 1)
                lm = fmaxf(lm, __shfl_xor(lm, xm, 32));
            const float mn = fmaxf(mrow[r], lm);
            alpha[r] = __expf(mrow[r] - mn);
            const float p0 = __expf(s[0][r] - mn);
            const float p1 = __expf(s[1][r] - mn);
            s[0][r] = p0; s[1][r] = p1;
            float rs = p0 + p1;
#pragma unroll
            for (int xm = 1; xm < 16; xm <<= 1)
                rs += __shfl_xor(rs, xm, 32);
            lrow[r] = lrow[r] * alpha[r] + rs;
            mrow[r] = mn;
        }
#pragma unroll
        for (int c = 0; c < 8; ++c)
#pragma unroll
            for (int r = 0; r < 8; ++r) o[c][r] *= alpha[r];

        // ---- transpose P (C-layout -> A-layout) via per-wave LDS ----
#pragma unroll
        for (int t = 0; t < 2; ++t)
#pragma unroll
            for (int r = 0; r < 8; ++r)
                myp[(r + 8 * grp) * 32 + t * 16 + lid] = (f16)s[t][r];
        wait_ds0();
        v16h pf = make_frag(myp + lid * 32 + koff, myp + lid * 32 + koff + 16);

        // ---- O += P @ V: B-fragments are contiguous b128 reads of V^T ----
#pragma unroll
        for (int c = 0; c < 8; ++c) {
            const f16* vb = &ldsVT[buf][(c * 16 + lid) * VTP + koff];
            o[c] = wmma_f16(pf, make_frag(vb, vb + 16), o[c]);
        }

        __syncthreads();   // everyone done reading `buf` before its reuse
        buf ^= 1;
    }

    // ---- epilogue: O /= l ----
    float inv[8];
#pragma unroll
    for (int r = 0; r < 8; ++r) inv[r] = 1.0f / lrow[r];
#pragma unroll
    for (int c = 0; c < 8; ++c)
#pragma unroll
        for (int r = 0; r < 8; ++r) {
            const size_t row = (size_t)(b * SEQ + m_base + r + 8 * grp);
            oh[row * (NH * HD) + hq * HD + c * 16 + lid] =
                (f16)(o[c][r] * inv[r]);
        }
}

// ---------------------------------------------------------------------------
// Host-side launch
// ---------------------------------------------------------------------------
extern "C" void kernel_launch(void* const* d_in, const int* in_sizes, int n_in,
                              void* d_out, int out_size, void* d_ws, size_t ws_size,
                              hipStream_t stream) {
    (void)in_sizes; (void)n_in; (void)out_size; (void)ws_size;

    const float* x    = (const float*)d_in[0];
    const float* wq   = (const float*)d_in[1];
    const float* wk   = (const float*)d_in[2];
    const float* wv   = (const float*)d_in[3];
    const float* wo   = (const float*)d_in[4];
    const float* cosb = (const float*)d_in[5];
    const float* sinb = (const float*)d_in[6];
    // d_in[7] = mask (causal, applied inline), d_in[8] = start_pos (0)

    char*  p   = (char*)d_ws;
    size_t off = 0;
    auto alloc = [&](size_t bytes) -> void* {
        void* r = p + off;
        off += (bytes + 255) & ~(size_t)255;
        return r;
    };
    f16*   xh  = (f16*)  alloc((size_t)MROWS * DIM  * 2);
    f16*   wqh = (f16*)  alloc((size_t)DIM   * DIM  * 2);
    f16*   wkh = (f16*)  alloc((size_t)KVDIM * DIM  * 2);
    f16*   wvh = (f16*)  alloc((size_t)KVDIM * DIM  * 2);
    f16*   woh = (f16*)  alloc((size_t)DIM   * DIM  * 2);
    float* qf  = (float*)alloc((size_t)MROWS * DIM  * 4);
    float* kf  = (float*)alloc((size_t)MROWS * KVDIM * 4);
    float* vf  = (float*)alloc((size_t)MROWS * KVDIM * 4);
    f16*   qh2 = (f16*)  alloc((size_t)MROWS * DIM  * 2);
    f16*   kh2 = (f16*)  alloc((size_t)MROWS * KVDIM * 2);
    f16*   vTt = (f16*)  alloc((size_t)MROWS * KVDIM * 2);  // (B,NKVH,HD,N)
    f16*   ah  = (f16*)  alloc((size_t)MROWS * DIM  * 2);

    const int nxd  = MROWS * DIM;      // 16,777,216
    const int nkvd = KVDIM * DIM;      // 4,194,304
    const int nkv  = MROWS * KVDIM;    // 4,194,304

    // 1) convert inputs to f16
    f32_to_f16_kernel<<<(nxd  + 255) / 256, 256, 0, stream>>>(x,  xh,  nxd);
    f32_to_f16_kernel<<<(nxd  + 255) / 256, 256, 0, stream>>>(wq, wqh, nxd);
    f32_to_f16_kernel<<<(nkvd + 255) / 256, 256, 0, stream>>>(wk, wkh, nkvd);
    f32_to_f16_kernel<<<(nkvd + 255) / 256, 256, 0, stream>>>(wv, wvh, nkvd);
    f32_to_f16_kernel<<<(nxd  + 255) / 256, 256, 0, stream>>>(wo, woh, nxd);

    // 2) projections: q = x@wq^T, k = x@wk^T, v = x@wv^T (f32 accum out)
    gemm_wmma_kernel<float><<<dim3(DIM / 64, MROWS / 128), 256, 0, stream>>>(
        xh, wqh, qf, MROWS, DIM, DIM);
    gemm_wmma_kernel<float><<<dim3(KVDIM / 64, MROWS / 128), 256, 0, stream>>>(
        xh, wkh, kf, MROWS, KVDIM, DIM);
    gemm_wmma_kernel<float><<<dim3(KVDIM / 64, MROWS / 128), 256, 0, stream>>>(
        xh, wvh, vf, MROWS, KVDIM, DIM);

    // 3) RoPE on q and k (writes f16); v transposed to (B,NKVH,HD,N) f16
    const int qtot = MROWS * NH * 64;
    const int ktot = MROWS * NKVH * 64;
    rope_kernel<<<(qtot + 255) / 256, 256, 0, stream>>>(qf, qh2, cosb, sinb, NH,  qtot);
    rope_kernel<<<(ktot + 255) / 256, 256, 0, stream>>>(kf, kh2, cosb, sinb, NKVH, ktot);
    transpose_v_kernel<<<(nkv + 255) / 256, 256, 0, stream>>>(vf, vTt, nkv);

    // 4) causal flash attention
    flash_attn_kernel<<<BATCH * NH * (SEQ / 64), 128, 0, stream>>>(qh2, kh2, vTt, ah);

    // 5) output projection: out = ah @ wo^T (f32 out)
    gemm_wmma_kernel<float><<<dim3(DIM / 64, MROWS / 128), 256, 0, stream>>>(
        ah, woh, (float*)d_out, MROWS, DIM, DIM);
}
